// VQVAE_24043226923942
// MI455X (gfx1250) — matrix operand order repaired and checked by reading, not compiled, not run on hardware
//
#include <hip/hip_runtime.h>
#include <hip/hip_bf16.h>
#include <stdint.h>
#include <stddef.h>

// ---------------------------------------------------------------------------
// Types for CDNA5 WMMA (gfx1250, wave32): v16bf A/B fragments, v8f C/D.
// ---------------------------------------------------------------------------
typedef __attribute__((ext_vector_type(16))) __bf16        v16bf;
typedef __attribute__((ext_vector_type(8)))  float         v8f;
typedef __attribute__((ext_vector_type(4)))  unsigned int  u32x4;   // POD 128-bit load

#define CF_TRANSPOSED 1
#define CF_RELU_IN    2
#define CF_RELU_OUT   4
#define CF_F32_OUT    8

__device__ __forceinline__ unsigned short f2bf_bits(float f) {
  union { float f; unsigned u; } v; v.f = f;
  unsigned r = v.u + 0x7FFFu + ((v.u >> 16) & 1u);   // round-to-nearest-even
  return (unsigned short)(r >> 16);
}
__device__ __forceinline__ float bfbits2f(unsigned short u) {
  union { unsigned u; float f; } t; t.u = (unsigned)u << 16; return t.f;
}

// 32-byte fragment viewed as two 128-bit loads / 16 halfwords / a v16bf operand.
union Frag32B { u32x4 q[2]; unsigned short h[16]; v16bf v; };

// Packed relu on 2x bf16 per dword: zero any halfword with the sign bit set.
__device__ __forceinline__ u32x4 relu_pk(u32x4 u) {
#pragma unroll
  for (int j = 0; j < 4; ++j) {
    unsigned m = ((u[j] >> 15) & 0x00010001u) * 0xFFFFu;
    u[j] &= ~m;
  }
  return u;
}

// ---------------------------------------------------------------------------
// Weight packing into GEMM order k = (kh*KW + kw)*IC + ic  (spatial-major,
// channel-minor): one 16-long K-run per lane = 16 consecutive channels at ONE
// spatial tap. wp rows are [OCpad][Kpad] bf16, zero padded.
// ---------------------------------------------------------------------------
__global__ void pack_w_kernel(const float* __restrict__ w, unsigned short* __restrict__ wp,
                              int OC, int IC, int KHW, int K, int Kpad, int total) {
  int i = blockIdx.x * 256 + threadIdx.x;
  if (i >= total) return;
  int k = i % Kpad, oc = i / Kpad;
  float v = 0.f;
  if (oc < OC && k < K) {
    int sp = k / IC;
    int ic = k - sp * IC;
    v = w[(size_t)oc * K + (size_t)ic * KHW + sp];       // OIHW source
  }
  wp[i] = f2bf_bits(v);
}

// ConvTranspose weights: torch layout (Cin, Cout, kh, kw).
__global__ void pack_wT_kernel(const float* __restrict__ w, unsigned short* __restrict__ wp,
                               int OC, int IC, int KHW, int K, int Kpad, int total) {
  int i = blockIdx.x * 256 + threadIdx.x;
  if (i >= total) return;
  int k = i % Kpad, oc = i / Kpad;
  float v = 0.f;
  if (oc < OC && k < K) {
    int sp = k / IC;
    int ic = k - sp * IC;
    v = w[((size_t)ic * OC + oc) * KHW + sp];
  }
  wp[i] = f2bf_bits(v);
}

// embed [64][512] fp32 -> ep [512][64] bf16 (d-contiguous per code), e2 = ||e||^2
__global__ void pack_embed_kernel(const float* __restrict__ e, unsigned short* __restrict__ ep,
                                  float* __restrict__ e2, int D, int NC) {
  int code = blockIdx.x * blockDim.x + threadIdx.x;
  if (code >= NC) return;
  float s = 0.f;
  for (int d = 0; d < D; ++d) {
    float v = e[(size_t)d * NC + code];
    ep[(size_t)code * D + d] = f2bf_bits(v);
    s += v * v;
  }
  e2[code] = s;
}

// fp32 NCHW -> bf16 NHWC (for the network input x).
__global__ void nchw_to_nhwc_kernel(const float* __restrict__ xin, unsigned short* __restrict__ xo,
                                    int C, int HW, int total) {
  int i = blockIdx.x * 256 + threadIdx.x;
  if (i >= total) return;
  int c = i % C;
  int t = i / C;
  int hw = t % HW;
  int b  = t / HW;
  xo[i] = f2bf_bits(xin[((size_t)b * C + c) * HW + hw]);
}

// ---------------------------------------------------------------------------
// Implicit-GEMM conv / conv-transpose over bf16 NHWC activations.
// Block = 4 independent waves (threadIdx.y = pixel-tile sub-index) to lift the
// 32-workgroup-per-WGP residency cap; each wave computes a 16*OCB(OC) x
// 16(pixel) tile with EXEC all-ones (WMMA requirement).
//  A (16x32 bf16): lane m=lane&15, half=lane>>4 -> two b128 weight loads.
//  B (32x16 bf16): lane n=lane&15 -> 16 consecutive channels at one tap:
//                  two b128 activation loads (IC%16==0 fast path; IC=2 generic).
//  C (16x16 f32): lanes 0-15 M=r, lanes 16-31 M=r+8; 8 consecutive OCs per
//                 (s,half) -> one packed b128 bf16 store (or f32 store, OC=1).
// B fragment is built ONCE per chunk and reused by OCB WMMAs.
// ---------------------------------------------------------------------------
template<int OCB, int KH, int KW, bool TR>
__global__ __launch_bounds__(128) void conv_wmma_kernel(
    const unsigned short* __restrict__ in, const unsigned short* __restrict__ wp,
    const float* __restrict__ bias, const unsigned short* __restrict__ addsrc,
    void* __restrict__ out,
    int IC, int IH, int IW, int OC, int OH, int OW,
    int stride, int pad, int K, int Kpad, int flags)
{
  const int lane = threadIdx.x;
  const int n    = lane & 15;
  const int half = lane >> 4;
  const int tile = blockIdx.x * 4 + threadIdx.y;   // per-wave pixel tile
  const int OHOW = OH * OW;
  const int p    = tile * 16 + n;
  const int b    = p / OHOW;
  const int rem  = p - b * OHOW;
  const int oh   = rem / OW;
  const int ow   = rem - oh * OW;
  const int ocBase = blockIdx.y * (16 * OCB);
  const bool relu_in = (flags & CF_RELU_IN) != 0;
  const int IHIW = IH * IW;
  const unsigned short* inb = in + (size_t)b * IHIW * IC;
  const bool icAligned = (IC & 15) == 0;

  // Running decomposition of this half's K-run base (kc + half*16) = spat*IC + icb.
  int spat = (half * 16) / IC;
  int icb  = (half * 16) - spat * IC;

  v8f acc[OCB];
#pragma unroll
  for (int s = 0; s < OCB; ++s) { v8f z = {}; acc[s] = z; }

  const unsigned short* wrow[OCB];
#pragma unroll
  for (int s = 0; s < OCB; ++s)
    wrow[s] = wp + (size_t)(ocBase + s * 16 + n) * Kpad;     // A row: m = lane&15

  for (int kc = 0; kc < Kpad; kc += 32) {
    Frag32B bfr;
    if (icAligned) {
      // ---- fast path: one spatial tap, 16 consecutive NHWC channels ----
      int kh = spat / KW, kw = spat - kh * KW;               // constexpr divisor
      int ih, iw; bool ok;
      if (!TR) {
        ih = oh * stride - pad + kh;
        iw = ow * stride - pad + kw;
        ok = ((unsigned)ih < (unsigned)IH) & ((unsigned)iw < (unsigned)IW);
      } else {
        int nh = oh + pad - kh, nw = ow + pad - kw;
        ih = nh / stride; iw = nw / stride;
        ok = (nh >= 0) & (nw >= 0) & (nh - ih * stride == 0) &
             (nw - iw * stride == 0) & (ih < IH) & (iw < IW);
      }
      ok = ok & ((kc + half * 16) < K);
      if (ok) {
        const unsigned short* src = inb + ((size_t)ih * IW + iw) * IC + icb;
        bfr.q[0] = *(const u32x4*)(src);
        bfr.q[1] = *(const u32x4*)(src + 8);
        if (relu_in) { bfr.q[0] = relu_pk(bfr.q[0]); bfr.q[1] = relu_pk(bfr.q[1]); }
      } else {
        bfr.q[0] = (u32x4)0u;
        bfr.q[1] = (u32x4)0u;
      }
    } else {
      // ---- generic path (first conv, IC=2): per-element decode ----
#pragma unroll
      for (int i = 0; i < 16; ++i) {
        int k = kc + half * 16 + i;
        unsigned short uv = 0;
        if (k < K) {
          int sp = k / IC;
          int kh = sp / KW, kw = sp - kh * KW;
          int ic = k - sp * IC;
          int ih, iw; bool ok;
          if (!TR) {
            ih = oh * stride - pad + kh;
            iw = ow * stride - pad + kw;
            ok = ((unsigned)ih < (unsigned)IH) & ((unsigned)iw < (unsigned)IW);
          } else {
            int nh = oh + pad - kh, nw = ow + pad - kw;
            ih = nh / stride; iw = nw / stride;
            ok = (nh >= 0) & (nw >= 0) & (nh - ih * stride == 0) &
                 (nw - iw * stride == 0) & (ih < IH) & (iw < IW);
          }
          if (ok) {
            uv = inb[((size_t)ih * IW + iw) * IC + ic];
            if (relu_in && (uv & 0x8000u)) uv = 0;
          }
        }
        bfr.h[i] = uv;
      }
    }
    // advance (spat, icb) by one chunk (32) for next iteration
    icb += 32;
    if (icb >= IC) { icb -= IC; ++spat; }
    if (icb >= IC) { icb -= IC; ++spat; }

    // OCB sub-tiles share bfr: 2x b128 weight loads + 1 WMMA each.
#pragma unroll
    for (int s = 0; s < OCB; ++s) {
      Frag32B a;
      a.q[0] = *(const u32x4*)(wrow[s] + kc + half * 16);
      a.q[1] = *(const u32x4*)(wrow[s] + kc + half * 16 + 8);
      acc[s] = __builtin_amdgcn_wmma_f32_16x16x32_bf16(false, a.v, false, bfr.v,
                                                       (short)0, acc[s], false, false);
    }
  }

  if (flags & CF_F32_OUT) {
    // Final conv: OC==1, fp32 NCHW output straight into d_out.
    if (half == 0)
      ((float*)out)[(size_t)b * OHOW + rem] = acc[0][0] + bias[0];
    return;
  }

  const size_t pixbase = (size_t)b * OHOW + rem;             // NHWC pixel index
#pragma unroll
  for (int s = 0; s < OCB; ++s) {
    int ocb0 = ocBase + s * 16 + half * 8;                   // 8 consecutive OCs
    u32x4 av;
    if (addsrc) av = *(const u32x4*)(addsrc + pixbase * OC + ocb0);
    float vals[8];
#pragma unroll
    for (int r = 0; r < 8; ++r) {
      float v = acc[s][r] + bias[ocb0 + r];
      if (addsrc)
        v += bfbits2f((unsigned short)((av[r >> 1] >> ((r & 1) * 16)) & 0xFFFFu));
      if (flags & CF_RELU_OUT) v = fmaxf(v, 0.f);
      vals[r] = v;
    }
    u32x4 o;
#pragma unroll
    for (int j = 0; j < 4; ++j)
      o[j] = (unsigned)f2bf_bits(vals[2 * j]) | ((unsigned)f2bf_bits(vals[2 * j + 1]) << 16);
    *(u32x4*)((unsigned short*)out + pixbase * OC + ocb0) = o;
  }
}

// ---------------------------------------------------------------------------
// VQ argmin: dist(pixel, code) = -2 f.e + ||e||^2 (||f||^2 argmin-invariant).
// f is bf16 [Npix][64] (NHWC), so A fragments are pure b128 loads.
// 4 waves per block, one 16-pixel tile per wave.
// ---------------------------------------------------------------------------
__global__ __launch_bounds__(128) void vq_argmin_kernel(
    const unsigned short* __restrict__ f, const unsigned short* __restrict__ ep,
    const float* __restrict__ e2, int* __restrict__ idx_out)
{
  const int lane = threadIdx.x;
  const int n    = lane & 15;
  const int half = lane >> 4;
  const int tile = blockIdx.x * 4 + threadIdx.y;
  const int p    = tile * 16 + n;            // A: m = lane&15 -> this pixel
  const unsigned short* xrow = f + (size_t)p * 64;

  Frag32B a0, a1;
  a0.q[0] = *(const u32x4*)(xrow + half * 16);
  a0.q[1] = *(const u32x4*)(xrow + half * 16 + 8);
  a1.q[0] = *(const u32x4*)(xrow + 32 + half * 16);
  a1.q[1] = *(const u32x4*)(xrow + 32 + half * 16 + 8);

  float bestv[8];
  int   besti[8];
#pragma unroll
  for (int r = 0; r < 8; ++r) { bestv[r] = 3.4e38f; besti[r] = 0; }

  for (int ct = 0; ct < 32; ++ct) {
    int code = ct * 16 + n;                  // B/C: n = lane&15 -> this code
    const unsigned short* row = ep + (size_t)code * 64;
    Frag32B b0, b1;
    b0.q[0] = *(const u32x4*)(row + half * 16);
    b0.q[1] = *(const u32x4*)(row + half * 16 + 8);
    b1.q[0] = *(const u32x4*)(row + 32 + half * 16);
    b1.q[1] = *(const u32x4*)(row + 32 + half * 16 + 8);
    v8f acc = {};
    acc = __builtin_amdgcn_wmma_f32_16x16x32_bf16(false, a0.v, false, b0.v, (short)0, acc, false, false);
    acc = __builtin_amdgcn_wmma_f32_16x16x32_bf16(false, a1.v, false, b1.v, (short)0, acc, false, false);
    float eq = e2[code];
#pragma unroll
    for (int r = 0; r < 8; ++r) {
      float d = eq - 2.0f * acc[r];
      if (d < bestv[r] || (d == bestv[r] && code < besti[r])) { bestv[r] = d; besti[r] = code; }
    }
  }

  // Reduce over the 16 lanes of each half (codes live across lanes; pixels stay).
#pragma unroll
  for (int m = 1; m <= 8; m <<= 1) {
#pragma unroll
    for (int r = 0; r < 8; ++r) {
      float ov = __shfl_xor(bestv[r], m, 32);
      int   oi = __shfl_xor(besti[r], m, 32);
      if (ov < bestv[r] || (ov == bestv[r] && oi < besti[r])) { bestv[r] = ov; besti[r] = oi; }
    }
  }
  if (n == 0) {
#pragma unroll
    for (int r = 0; r < 8; ++r)
      idx_out[tile * 16 + r + half * 8] = besti[r];          // C: M = r + half*8
  }
}

// Gather q = embed[:, idx] as bf16 NHWC, per-pixel squared error to ps.
__global__ void vq_gather_kernel(const unsigned short* __restrict__ x,
                                 const float* __restrict__ embed,
                                 const int* __restrict__ idx,
                                 unsigned short* __restrict__ q,
                                 float* __restrict__ ps, int Npix)
{
  int p = blockIdx.x * blockDim.x + threadIdx.x;
  if (p >= Npix) return;
  int code = idx[p];
  float s = 0.f;
  const unsigned short* xr = x + (size_t)p * 64;
  unsigned short* qr = q + (size_t)p * 64;
  for (int d = 0; d < 64; ++d) {
    float e = embed[(size_t)d * 512 + code];
    float df = e - bfbits2f(xr[d]);
    qr[d] = f2bf_bits(e);
    s += df * df;
  }
  ps[p] = s;
}

// NHWC channel-offset copy (for the two concats), bf16.
__global__ void copy_channels_kernel(const unsigned short* __restrict__ src,
                                     unsigned short* __restrict__ dst,
                                     int Cs, int Cd, int coff, int total)
{
  int i = blockIdx.x * blockDim.x + threadIdx.x;
  if (i >= total) return;
  int c   = i % Cs;
  int bhw = i / Cs;
  dst[(size_t)bhw * Cd + coff + c] = src[i];
}

// Deterministic single-block reduction: out = sa*sum(a) + sb*sum(b).
__global__ void reduce_diff_kernel(const float* __restrict__ a, int na, float sa,
                                   const float* __restrict__ b, int nb, float sb,
                                   float* __restrict__ out)
{
  __shared__ float sh[256];
  float acc = 0.f;
  for (int i = threadIdx.x; i < na; i += 256) acc += a[i];
  acc *= sa;
  float accb = 0.f;
  for (int i = threadIdx.x; i < nb; i += 256) accb += b[i];
  acc += accb * sb;
  sh[threadIdx.x] = acc;
  __syncthreads();
  for (int st = 128; st > 0; st >>= 1) {
    if (threadIdx.x < st) sh[threadIdx.x] += sh[threadIdx.x + st];
    __syncthreads();
  }
  if (threadIdx.x == 0) out[0] = sh[0];
}

// ---------------------------------------------------------------------------
// Host orchestration
// ---------------------------------------------------------------------------
namespace {

struct Bump {
  char* base; size_t off, cap;
  void* take(size_t bytes) {
    size_t a = (bytes + 255) & ~(size_t)255;
    void* r = base + off; off += a; return r;
  }
};

inline int cdiv(int a, int b) { return (a + b - 1) / b; }

struct ResP { const float *w1, *b1, *w2, *b2; };

template<int KH, int KW, bool TR>
void conv_dispatch(hipStream_t s, const unsigned short* in, const unsigned short* wp,
                   const float* bias, const unsigned short* addsrc, void* out,
                   int IC, int IH, int IW, int OC, int OH, int OW,
                   int stride, int pad, int K, int Kpad, int flags,
                   int OCp, int Npix)
{
  dim3 blk(32, 4);
  if (OCp % 64 == 0)
    conv_wmma_kernel<4, KH, KW, TR><<<dim3(Npix / 64, OCp / 64), blk, 0, s>>>(
        in, wp, bias, addsrc, out, IC, IH, IW, OC, OH, OW, stride, pad, K, Kpad, flags);
  else if (OCp % 32 == 0)
    conv_wmma_kernel<2, KH, KW, TR><<<dim3(Npix / 64, OCp / 32), blk, 0, s>>>(
        in, wp, bias, addsrc, out, IC, IH, IW, OC, OH, OW, stride, pad, K, Kpad, flags);
  else
    conv_wmma_kernel<1, KH, KW, TR><<<dim3(Npix / 64, OCp / 16), blk, 0, s>>>(
        in, wp, bias, addsrc, out, IC, IH, IW, OC, OH, OW, stride, pad, K, Kpad, flags);
}

void run_conv(hipStream_t s, Bump& ws,
              const unsigned short* in, const float* w, const float* bias,
              const unsigned short* addsrc, void* out, int B, int IC, int IH, int IW, int OC,
              int KH, int KW, int stride, int pad, int flags)
{
  int K = IC * KH * KW;
  int Kpad = (K + 31) & ~31;
  int OCp  = (OC + 15) & ~15;
  int total = OCp * Kpad;
  unsigned short* wp = (unsigned short*)ws.take((size_t)total * 2);
  if (flags & CF_TRANSPOSED)
    pack_wT_kernel<<<cdiv(total, 256), 256, 0, s>>>(w, wp, OC, IC, KH * KW, K, Kpad, total);
  else
    pack_w_kernel<<<cdiv(total, 256), 256, 0, s>>>(w, wp, OC, IC, KH * KW, K, Kpad, total);

  int OH, OW;
  if (flags & CF_TRANSPOSED) { OH = (IH - 1) * stride - 2 * pad + KH; OW = (IW - 1) * stride - 2 * pad + KW; }
  else                       { OH = (IH + 2 * pad - KH) / stride + 1; OW = (IW + 2 * pad - KW) / stride + 1; }
  int Npix = B * OH * OW;

  if (flags & CF_TRANSPOSED)
    conv_dispatch<4, 4, true >(s, in, wp, bias, addsrc, out, IC, IH, IW, OC, OH, OW, stride, pad, K, Kpad, flags, OCp, Npix);
  else if (KH == 3)
    conv_dispatch<3, 3, false>(s, in, wp, bias, addsrc, out, IC, IH, IW, OC, OH, OW, stride, pad, K, Kpad, flags, OCp, Npix);
  else if (KH == 1)
    conv_dispatch<1, 1, false>(s, in, wp, bias, addsrc, out, IC, IH, IW, OC, OH, OW, stride, pad, K, Kpad, flags, OCp, Npix);
  else
    conv_dispatch<4, 4, false>(s, in, wp, bias, addsrc, out, IC, IH, IW, OC, OH, OW, stride, pad, K, Kpad, flags, OCp, Npix);
}

void run_vq(hipStream_t s, Bump& ws, const unsigned short* f, const float* embed,
            unsigned short* q, float* ps, int* idx, int Npix)
{
  unsigned short* ep = (unsigned short*)ws.take((size_t)512 * 64 * 2);
  float* e2 = (float*)ws.take(512 * 4);
  pack_embed_kernel<<<2, 256, 0, s>>>(embed, ep, e2, 64, 512);
  vq_argmin_kernel<<<Npix / 64, dim3(32, 4), 0, s>>>(f, ep, e2, idx);
  vq_gather_kernel<<<cdiv(Npix, 256), 256, 0, s>>>(f, embed, idx, q, ps, Npix);
}

} // namespace

extern "C" void kernel_launch(void* const* d_in, const int* in_sizes, int n_in,
                              void* d_out, int out_size, void* d_ws, size_t ws_size,
                              hipStream_t stream)
{
  (void)in_sizes; (void)n_in; (void)out_size;
  int ii = 0;
  auto P = [&]() { return (const float*)d_in[ii++]; };
  auto takeRes = [&]() { ResP r; r.w1 = P(); r.b1 = P(); r.w2 = P(); r.b2 = P(); return r; };

  const float* x = P();
  // enc_b
  const float *eb_w1 = P(), *eb_b1 = P(), *eb_w2 = P(), *eb_b2 = P(), *eb_w3 = P(), *eb_b3 = P();
  ResP eb_r1 = takeRes(), eb_r2 = takeRes();
  // enc_t
  const float *et_w1 = P(), *et_b1 = P(), *et_w2 = P(), *et_b2 = P();
  ResP et_r1 = takeRes(), et_r2 = takeRes();
  const float *qct_w = P(), *qct_b = P();
  const float *embed_t = P();
  // dec_t
  const float *dt_w1 = P(), *dt_b1 = P();
  ResP dt_r1 = takeRes(), dt_r2 = takeRes();
  const float *dt_wt = P(), *dt_bt = P();
  const float *qcb_w = P(), *qcb_b = P();
  const float *embed_b = P();
  const float *up_w = P(), *up_b = P();
  // dec
  const float *dc_w1 = P(), *dc_b1 = P();
  ResP dc_r1 = takeRes(), dc_r2 = takeRes();
  const float *dc_wt1 = P(), *dc_bt1 = P(), *dc_wt2 = P(), *dc_bt2 = P();

  typedef unsigned short bfu;
  Bump ws{(char*)d_ws, 0, ws_size};
  bfu* XH     = (bfu*)ws.take((size_t)16 * 256 * 256 * 2 * 2);    // x as bf16 NHWC
  bfu* H1     = (bfu*)ws.take((size_t)16 * 128 * 128 * 64 * 2);   // also dec t1
  bfu* A      = (bfu*)ws.take((size_t)16 * 64 * 64 * 128 * 2);    // also QC
  bfu* Bb     = (bfu*)ws.take((size_t)16 * 64 * 64 * 128 * 2);
  bfu* ENCB   = (bfu*)ws.take((size_t)16 * 64 * 64 * 128 * 2);
  bfu* RH     = (bfu*)ws.take((size_t)16 * 64 * 64 * 32 * 2);     // res hidden (max)
  bfu* S      = (bfu*)ws.take((size_t)16 * 32 * 32 * 64 * 2);
  bfu* U      = (bfu*)ws.take((size_t)16 * 32 * 32 * 128 * 2);
  bfu* V      = (bfu*)ws.take((size_t)16 * 32 * 32 * 128 * 2);
  bfu* ENCT   = (bfu*)ws.take((size_t)16 * 32 * 32 * 128 * 2);
  bfu* QT     = (bfu*)ws.take((size_t)16 * 32 * 32 * 64 * 2);
  bfu* QUANTT = (bfu*)ws.take((size_t)16 * 32 * 32 * 64 * 2);
  bfu* DECT   = (bfu*)ws.take((size_t)16 * 64 * 64 * 64 * 2);
  bfu* CAT    = (bfu*)ws.take((size_t)16 * 64 * 64 * 192 * 2);
  bfu* QB     = (bfu*)ws.take((size_t)16 * 64 * 64 * 64 * 2);
  bfu* QUANTB = (bfu*)ws.take((size_t)16 * 64 * 64 * 64 * 2);
  bfu* UPT    = (bfu*)ws.take((size_t)16 * 64 * 64 * 64 * 2);
  int* IDX_T  = (int*)ws.take(16384 * 4);
  int* IDX_B  = (int*)ws.take(65536 * 4);
  float* PS_T = (float*)ws.take(16384 * 4);
  float* PS_B = (float*)ws.take(65536 * 4);

  hipStream_t s = stream;
  // ReLU lives at producers (CF_RELU_OUT); CF_RELU_IN only where the raw
  // tensor is also needed (residual 3x3 reading the skip source).
  auto res_block = [&](const bfu* xin, const ResP& rp, bfu* hidden, bfu* outb,
                       int H, int W, bool relu_out) {
    run_conv(s, ws, xin, rp.w1, rp.b1, nullptr, hidden, 16, 128, H, W, 32, 3, 3, 1, 1,
             CF_RELU_IN | CF_RELU_OUT);
    run_conv(s, ws, hidden, rp.w2, rp.b2, xin, outb, 16, 32, H, W, 128, 1, 1, 1, 0,
             relu_out ? CF_RELU_OUT : 0);
  };

  // ---- input: fp32 NCHW -> bf16 NHWC ----
  nchw_to_nhwc_kernel<<<cdiv(16 * 65536 * 2, 256), 256, 0, s>>>(x, XH, 2, 65536, 16 * 65536 * 2);

  // ---- encoder4 (enc_b): XH [16,256,256,2] -> ENCB [16,64,64,128] (relu'd) ----
  run_conv(s, ws, XH,  eb_w1, eb_b1, nullptr, H1, 16, 2,   256, 256, 64,  4, 4, 2, 1, CF_RELU_OUT);
  run_conv(s, ws, H1,  eb_w2, eb_b2, nullptr, A,  16, 64,  128, 128, 128, 4, 4, 2, 1, CF_RELU_OUT);
  run_conv(s, ws, A,   eb_w3, eb_b3, nullptr, Bb, 16, 128, 64,  64,  128, 3, 3, 1, 1, 0);
  res_block(Bb, eb_r1, RH, A, 64, 64, false);
  res_block(A,  eb_r2, RH, ENCB, 64, 64, true);

  // ---- encoder2 (enc_t): ENCB -> ENCT [16,32,32,128] (relu'd) ----
  run_conv(s, ws, ENCB, et_w1, et_b1, nullptr, S, 16, 128, 64, 64, 64,  4, 4, 2, 1, CF_RELU_OUT);
  run_conv(s, ws, S,    et_w2, et_b2, nullptr, U, 16, 64,  32, 32, 128, 3, 3, 1, 1, 0);
  res_block(U, et_r1, RH, V, 32, 32, false);
  res_block(V, et_r2, RH, ENCT, 32, 32, true);

  // ---- top quantize ----
  run_conv(s, ws, ENCT, qct_w, qct_b, nullptr, QT, 16, 128, 32, 32, 64, 1, 1, 1, 0, 0);
  run_vq(s, ws, QT, embed_t, QUANTT, PS_T, IDX_T, 16384);

  // ---- decoder2 (dec_t): QUANTT -> DECT [16,64,64,64] ----
  run_conv(s, ws, QUANTT, dt_w1, dt_b1, nullptr, U, 16, 64, 32, 32, 128, 3, 3, 1, 1, 0);
  res_block(U, dt_r1, RH, V, 32, 32, false);
  res_block(V, dt_r2, RH, U, 32, 32, true);
  run_conv(s, ws, U, dt_wt, dt_bt, nullptr, DECT, 16, 128, 32, 32, 64, 4, 4, 2, 1, CF_TRANSPOSED);

  // ---- cat_b = [DECT ; ENCB] (NHWC channel concat) -> bottom quantize ----
  copy_channels_kernel<<<cdiv(16 * 4096 * 64, 256), 256, 0, s>>>(DECT, CAT, 64, 192, 0, 16 * 4096 * 64);
  copy_channels_kernel<<<cdiv(16 * 4096 * 128, 256), 256, 0, s>>>(ENCB, CAT, 128, 192, 64, 16 * 4096 * 128);
  run_conv(s, ws, CAT, qcb_w, qcb_b, nullptr, QB, 16, 192, 64, 64, 64, 1, 1, 1, 0, 0);
  run_vq(s, ws, QB, embed_b, QUANTB, PS_B, IDX_B, 65536);

  // ---- up_t = convT(QUANTT); quant = [UPT ; QUANTB] -> A (as QC) ----
  run_conv(s, ws, QUANTT, up_w, up_b, nullptr, UPT, 16, 64, 32, 32, 64, 4, 4, 2, 1, CF_TRANSPOSED);
  copy_channels_kernel<<<cdiv(16 * 4096 * 64, 256), 256, 0, s>>>(UPT, A, 64, 128, 0, 16 * 4096 * 64);
  copy_channels_kernel<<<cdiv(16 * 4096 * 64, 256), 256, 0, s>>>(QUANTB, A, 64, 128, 64, 16 * 4096 * 64);

  // ---- decoder4: A -> d_out [16,1,256,256] fp32 ----
  float* OUT = (float*)d_out;
  run_conv(s, ws, A, dc_w1, dc_b1, nullptr, Bb, 16, 128, 64, 64, 128, 3, 3, 1, 1, 0);
  res_block(Bb, dc_r1, RH, A, 64, 64, false);
  res_block(A,  dc_r2, RH, Bb, 64, 64, true);
  run_conv(s, ws, Bb, dc_wt1, dc_bt1, nullptr, H1, 16, 128, 64, 64, 64, 4, 4, 2, 1,
           CF_TRANSPOSED | CF_RELU_OUT);
  run_conv(s, ws, H1, dc_wt2, dc_bt2, nullptr, OUT, 16, 64, 128, 128, 1, 4, 4, 2, 1,
           CF_TRANSPOSED | CF_F32_OUT);

  // ---- diff = mean_t + mean_b, deterministic reduction into d_out tail ----
  reduce_diff_kernel<<<1, 256, 0, s>>>(PS_T, 16384, 1.f / (16384.f * 64.f),
                                       PS_B, 65536, 1.f / (65536.f * 64.f),
                                       OUT + (size_t)16 * 256 * 256);
}
